// Model_83760452207454
// MI455X (gfx1250) — compile-verified, matrix operand order checked by reading, not run on hardware
//
#include <hip/hip_runtime.h>
#include <hip/hip_bf16.h>

typedef __attribute__((ext_vector_type(16))) _Float16 v16h;
typedef __attribute__((ext_vector_type(8)))  _Float16 v8h;
typedef __attribute__((ext_vector_type(8)))  float    v8f;

#define BB 8
#define LL 2048
#define MROWS (BB*LL)          // 16384
#define DMODEL 256
#define DINNER 512
#define DSTATE 16
#define DTRANK 16
#define PRED 96
#define COUT 64

// ---------------------------------------------------------------------------
// Pack concat(x_enc, x_mark) into f16 [M, 96] (K=68 zero-padded to 96)
// ---------------------------------------------------------------------------
__global__ void pack_input(const float* __restrict__ xe, const float* __restrict__ xm,
                           _Float16* __restrict__ Xp) {
  int idx = blockIdx.x * blockDim.x + threadIdx.x;
  if (idx >= MROWS * 96) return;
  int m = idx / 96, c = idx % 96;
  float v = 0.0f;
  if (c < 64)      v = xe[m * 64 + c];
  else if (c < 68) v = xm[m * 4 + (c - 64)];
  Xp[idx] = (_Float16)v;
}

// ---------------------------------------------------------------------------
// Weight [N,K] f32 -> fragment-packed f16 B operand.
// Packed layout: Bp[((kt*tilesN + nt)*32 + lane)*16 + e], where
//   n  = nt*16 + (lane&15),  hi = lane>>4
//   e<8 : k = kt*32 + hi*8 + e          (VGPRs 0..3 of the fragment)
//   e>=8: k = kt*32 + 16 + hi*8 + (e-8) (VGPRs 4..7)
// so the GEMM wave loads its whole B fragment as one contiguous 32B v16h.
// ---------------------------------------------------------------------------
__device__ __forceinline__ void frag_decode(int idx, int tilesN,
                                            int& k, int& n) {
  int e    = idx & 15;
  int lane = (idx >> 4) & 31;
  int rem  = idx >> 9;
  int nt   = rem % tilesN;
  int kt   = rem / tilesN;
  int hi   = lane >> 4;
  n = nt * 16 + (lane & 15);
  k = kt * 32 + ((e < 8) ? 0 : 16) + hi * 8 + (e & 7);
}

__global__ void wconv_pack(const float* __restrict__ W, _Float16* __restrict__ Bp,
                           int N, int K, int Kpad) {
  int idx = blockIdx.x * blockDim.x + threadIdx.x;
  if (idx >= Kpad * N) return;
  int k, n;
  frag_decode(idx, N >> 4, k, n);
  Bp[idx] = (_Float16)((k < K) ? W[n * K + k] : 0.0f);
}

// Combined output weight Wc[n,k] = sum_r W_fc[n,r]*W_out[r,k], packed directly.
__global__ void wcomb_pack(const float* __restrict__ Wfc, const float* __restrict__ Wout,
                           _Float16* __restrict__ Bp) {
  int idx = blockIdx.x * blockDim.x + threadIdx.x;
  if (idx >= DINNER * COUT) return;
  int k, n;
  frag_decode(idx, COUT >> 4, k, n);
  float acc = 0.0f;
  for (int r = 0; r < DMODEL; ++r)
    acc += Wfc[n * DMODEL + r] * Wout[r * DINNER + k];
  Bp[idx] = (_Float16)acc;
}

// ---------------------------------------------------------------------------
// WMMA GEMM: C[M,N](f32) = A[M,K](f16) * Bp(packed f16) (+bias[n]).
// One wave computes a 64x16 supertile: 4 accumulators share each B fragment
// (4 v_wmma per K-step).  A fragments are two contiguous 16B runs per lane;
// B fragments are one contiguous 32B run (pre-swizzled).
// ---------------------------------------------------------------------------
__global__ void wmma_gemm(const _Float16* __restrict__ A, const _Float16* __restrict__ Bp,
                          float* __restrict__ C, _Float16* __restrict__ Ch,
                          const float* __restrict__ bias, int M, int N, int K) {
  int lane = threadIdx.x & 31;
  int wave = threadIdx.x >> 5;
  int tilesN = N >> 4;
  int totalTiles = (M >> 6) * tilesN;                 // 64-row supertiles
  int tile = blockIdx.x * (blockDim.x >> 5) + wave;   // wave-uniform
  if (tile >= totalTiles) return;                     // whole-wave exit
  int tm = tile / tilesN;
  int tn = tile % tilesN;
  int hi  = lane >> 4;
  int lid = lane & 15;
  int n  = tn * 16 + lid;
  int m0 = tm * 64 + lid;

  v8f acc[4];
  float bv = bias ? bias[n] : 0.0f;
#pragma unroll
  for (int s = 0; s < 4; ++s)
#pragma unroll
    for (int r = 0; r < 8; ++r) acc[s][r] = bv;

  const int ktiles = K >> 5;
  for (int kt = 0; kt < ktiles; ++kt) {
    // whole B fragment: one contiguous 32-byte load
    v16h b = *(const v16h*)(Bp + (((size_t)kt * tilesN + tn) * 32 + lane) * 16);
    const _Float16* Ab = A + (size_t)m0 * K + (kt << 5) + hi * 8;
#pragma unroll
    for (int s = 0; s < 4; ++s) {
      const _Float16* Ar = Ab + (size_t)s * 16 * K;
      v8h alo = *(const v8h*)(Ar);        // K[hi*8 .. hi*8+7]
      v8h ahi = *(const v8h*)(Ar + 16);   // K[16+hi*8 .. 16+hi*8+7]
      v16h a = __builtin_shufflevector(alo, ahi,
                 0,1,2,3,4,5,6,7,8,9,10,11,12,13,14,15);
      acc[s] = __builtin_amdgcn_wmma_f32_16x16x32_f16(false, a, false, b,
                                                      (short)0, acc[s], false, false);
    }
  }

  if (Ch) {
#pragma unroll
    for (int s = 0; s < 4; ++s)
#pragma unroll
      for (int r = 0; r < 8; ++r) {
        int mr = tm * 64 + s * 16 + r + hi * 8;
        float v = acc[s][r];
        C[(size_t)mr * N + n]  = v;
        Ch[(size_t)mr * N + n] = (_Float16)v;
      }
  } else {
#pragma unroll
    for (int s = 0; s < 4; ++s)
#pragma unroll
      for (int r = 0; r < 8; ++r) {
        int mr = tm * 64 + s * 16 + r + hi * 8;
        C[(size_t)mr * N + n] = acc[s][r];
      }
  }
}

// ---------------------------------------------------------------------------
// Causal depthwise conv1d (k=4) + bias + SiLU on xp = XZ[:, :512].
// ---------------------------------------------------------------------------
__global__ void conv_silu(const float* __restrict__ XZ, const float* __restrict__ cw,
                          const float* __restrict__ cb,
                          float* __restrict__ xc, _Float16* __restrict__ xch) {
  int idx = blockIdx.x * blockDim.x + threadIdx.x;
  if (idx >= MROWS * DINNER) return;
  int d = idx % DINNER;
  int m = idx / DINNER;
  int t = m % LL;
  float acc = cb[d];
#pragma unroll
  for (int j = 0; j < 4; ++j) {
    int tt = t - 3 + j;
    if (tt >= 0) acc += cw[d * 4 + j] * XZ[(size_t)(m - 3 + j) * 1024 + d];
  }
  float s = acc / (1.0f + __expf(-acc));   // SiLU
  xc[idx]  = s;
  xch[idx] = (_Float16)s;
}

// ---------------------------------------------------------------------------
// dt = softplus(x_db[:, :16] @ W_dt.T + b_dt).
// ---------------------------------------------------------------------------
__global__ void dt_softplus(const float* __restrict__ xdb, const float* __restrict__ Wdt,
                            const float* __restrict__ bdt, float* __restrict__ dt) {
  int idx = blockIdx.x * blockDim.x + threadIdx.x;
  if (idx >= MROWS * DINNER) return;
  int d = idx % DINNER, m = idx / DINNER;
  float acc = bdt[d];
#pragma unroll
  for (int r = 0; r < DTRANK; ++r)
    acc += xdb[m * 48 + r] * Wdt[d * DTRANK + r];
  dt[idx] = (acc > 20.0f) ? acc : log1pf(__expf(acc));
}

// ---------------------------------------------------------------------------
// Selective scan: one thread per (b, d, n); dA/dBu recomputed on the fly.
// y reduced over the 16 states with shfl_xor, only for the last PRED steps.
// ---------------------------------------------------------------------------
__global__ void ssm_scan(const float* __restrict__ dt, const float* __restrict__ xc,
                         const float* __restrict__ xdb, const float* __restrict__ XZ,
                         const float* __restrict__ A_log, const float* __restrict__ Dp,
                         _Float16* __restrict__ Ys) {
  int g = blockIdx.x * blockDim.x + threadIdx.x;   // 65536 threads
  int n = g & 15;
  int p = g >> 4;
  int b = p / DINNER;
  int d = p % DINNER;

  float a  = -__expf(A_log[d * DSTATE + n]);
  float Dv = Dp[d];
  float h  = 0.0f;
  const int base = b * LL;

  for (int t = 0; t < LL; ++t) {
    int m = base + t;
    float dtv = dt[(size_t)m * DINNER + d];
    float xcv = xc[(size_t)m * DINNER + d];
    float Bn  = xdb[m * 48 + DTRANK + n];
    float dA  = __expf(dtv * a);
    h = dA * h + dtv * Bn * xcv;
    if (t >= LL - PRED) {
      float Cn = xdb[m * 48 + DTRANK + DSTATE + n];
      float y  = h * Cn;
      y += __shfl_xor(y, 8);
      y += __shfl_xor(y, 4);
      y += __shfl_xor(y, 2);
      y += __shfl_xor(y, 1);
      if (n == 0) {
        float zv = XZ[(size_t)m * 1024 + DINNER + d];
        y = (y + xcv * Dv) * (zv / (1.0f + __expf(-zv)));
        Ys[((size_t)b * PRED + (t - (LL - PRED))) * DINNER + d] = (_Float16)y;
      }
    }
  }
}

// ---------------------------------------------------------------------------
extern "C" void kernel_launch(void* const* d_in, const int* in_sizes, int n_in,
                              void* d_out, int out_size, void* d_ws, size_t ws_size,
                              hipStream_t stream) {
  const float* x_enc  = (const float*)d_in[0];
  const float* x_mark = (const float*)d_in[1];
  const float* W_it   = (const float*)d_in[4];
  const float* b_it   = (const float*)d_in[5];
  const float* W_in   = (const float*)d_in[6];
  const float* conv_w = (const float*)d_in[7];
  const float* conv_b = (const float*)d_in[8];
  const float* W_x    = (const float*)d_in[9];
  const float* W_dt   = (const float*)d_in[10];
  const float* b_dt   = (const float*)d_in[11];
  const float* A_log  = (const float*)d_in[12];
  const float* Dp     = (const float*)d_in[13];
  const float* W_out  = (const float*)d_in[14];
  const float* W_fc   = (const float*)d_in[15];
  const float* b_fc   = (const float*)d_in[16];
  float* out = (float*)d_out;                       // [8, 96, 64] f32

  char* ws = (char*)d_ws;
  size_t off = 0;
  auto carve = [&](size_t bytes) -> char* {
    char* p = ws + off;
    off = (off + bytes + 255) & ~(size_t)255;
    return p;
  };
  _Float16* Xpad  = (_Float16*)carve((size_t)MROWS * 96 * 2);
  _Float16* WitB  = (_Float16*)carve((size_t)96 * DMODEL * 2);
  _Float16* WinB  = (_Float16*)carve((size_t)DMODEL * (2 * DINNER) * 2);
  _Float16* WxB   = (_Float16*)carve((size_t)DINNER * 48 * 2);
  _Float16* WcB   = (_Float16*)carve((size_t)DINNER * COUT * 2);
  float*    X1    = (float*)   carve((size_t)MROWS * DMODEL * 4);
  _Float16* X1h   = (_Float16*)carve((size_t)MROWS * DMODEL * 2);
  float*    XZ    = (float*)   carve((size_t)MROWS * 2 * DINNER * 4);
  float*    xcf   = (float*)   carve((size_t)MROWS * DINNER * 4);
  _Float16* xch   = (_Float16*)carve((size_t)MROWS * DINNER * 2);
  float*    xdb   = (float*)   carve((size_t)MROWS * 48 * 4);
  float*    dtb   = (float*)   carve((size_t)MROWS * DINNER * 4);
  _Float16* Ysh   = (_Float16*)carve((size_t)BB * PRED * DINNER * 2);
  (void)ws_size; (void)in_sizes; (void)n_in; (void)out_size;

  const int TPB = 256;
  auto blocks = [](long long nElems, int tpb) {
    return (unsigned)((nElems + tpb - 1) / tpb);
  };

  // 1) pack input (zero-padded K)
  pack_input<<<blocks((long long)MROWS * 96, TPB), TPB, 0, stream>>>(x_enc, x_mark, Xpad);

  // 2) weight conversions -> fragment-packed B operands
  wconv_pack<<<blocks(96LL * DMODEL, TPB), TPB, 0, stream>>>(W_it, WitB, DMODEL, 68, 96);
  wconv_pack<<<blocks((long long)DMODEL * 2 * DINNER, TPB), TPB, 0, stream>>>(W_in, WinB, 2 * DINNER, DMODEL, DMODEL);
  wconv_pack<<<blocks((long long)DINNER * 48, TPB), TPB, 0, stream>>>(W_x, WxB, 48, DINNER, DINNER);
  wcomb_pack<<<blocks((long long)DINNER * COUT, TPB), TPB, 0, stream>>>(W_fc, W_out, WcB);

  // 3) X1 = Xpad @ WitB + b_it   (M=16384, N=256, K=96)  -> f32 + f16
  {
    int tiles = (MROWS / 64) * (DMODEL / 16);
    wmma_gemm<<<blocks(tiles, 4), 128, 0, stream>>>(Xpad, WitB, X1, X1h, b_it,
                                                    MROWS, DMODEL, 96);
  }
  // 4) XZ = X1 @ WinB            (M=16384, N=1024, K=256)
  {
    int tiles = (MROWS / 64) * ((2 * DINNER) / 16);
    wmma_gemm<<<blocks(tiles, 4), 128, 0, stream>>>(X1h, WinB, XZ, nullptr, nullptr,
                                                    MROWS, 2 * DINNER, DMODEL);
  }
  // 5) xc = silu(causal_conv(xp) + conv_b)
  conv_silu<<<blocks((long long)MROWS * DINNER, TPB), TPB, 0, stream>>>(XZ, conv_w, conv_b, xcf, xch);

  // 6) x_db = xc @ WxB           (M=16384, N=48, K=512)
  {
    int tiles = (MROWS / 64) * (48 / 16);
    wmma_gemm<<<blocks(tiles, 4), 128, 0, stream>>>(xch, WxB, xdb, nullptr, nullptr,
                                                    MROWS, 48, DINNER);
  }
  // 7) dt = softplus(x_db[:, :16] @ W_dt.T + b_dt)
  dt_softplus<<<blocks((long long)MROWS * DINNER, TPB), TPB, 0, stream>>>(xdb, W_dt, b_dt, dtb);

  // 8) selective scan -> gated y slice (last 96 steps) as f16
  ssm_scan<<<blocks((long long)BB * DINNER * DSTATE, TPB), TPB, 0, stream>>>(
      dtb, xcf, xdb, XZ, A_log, Dp, Ysh);

  // 9) out = Ys @ (W_fc @ W_out).T + b_fc   (M=768, N=64, K=512)
  {
    int tiles = ((BB * PRED) / 64) * (COUT / 16);
    wmma_gemm<<<blocks(tiles, 4), 128, 0, stream>>>(Ysh, WcB, out, nullptr, b_fc,
                                                    BB * PRED, COUT, DINNER);
  }
}
